// OBM_GENConv_54571854463275
// MI455X (gfx1250) — matrix operand order repaired and checked by reading, not compiled, not run on hardware
//
#include <hip/hip_runtime.h>
#include <hip/hip_bf16.h>

// ---------------------------------------------------------------------------
// GENConv (2 layers) for MI455X / gfx1250.
// Dense math: v_wmma_f32_16x16x32_bf16 (bf16 in, f32 acc).
// Edge slabs staged via Tensor Data Mover (tensor_load_to_lds + tensorcnt).
// Softmax aggregation: two fused edge passes, L2-resident per-node atomics.
// ---------------------------------------------------------------------------

typedef __attribute__((ext_vector_type(16))) __bf16 v16bf;
typedef __attribute__((ext_vector_type(8)))  float  v8f;
typedef unsigned int u32x4 __attribute__((ext_vector_type(4)));
typedef int          i32x8 __attribute__((ext_vector_type(8)));
typedef int          i32x4 __attribute__((ext_vector_type(4)));

__device__ __forceinline__ __bf16 f2bf(float f) {
    union { float f; unsigned int u; } c; c.f = f;
    unsigned int u = c.u + 0x7FFFu + ((c.u >> 16) & 1u);   // round-nearest-even
    union { unsigned short s; __bf16 b; } r; r.s = (unsigned short)(u >> 16);
    return r.b;
}

__device__ __forceinline__ unsigned int pk_bf16x2(float lo, float hi) {
    union { float f; unsigned int u; } a, b; a.f = lo; b.f = hi;
    unsigned int ua = a.u + 0x7FFFu + ((a.u >> 16) & 1u);
    unsigned int ub = b.u + 0x7FFFu + ((b.u >> 16) & 1u);
    return (ua >> 16) | (ub & 0xFFFF0000u);
}

union FragBF { v16bf v; unsigned int u[8]; };

// A fragment from a bf16 [16][stride] LDS tile (ISA 7.12.2 16-bit A layout):
// lane m=l&15, half=l>>4; VGPR v: K = (v>>2)*16 + half*8 + (v&3)*2 (+pair).
__device__ __forceinline__ v16bf load_fragA(const __bf16* base, int stride, int kc, int lane) {
    FragBF f;
    const int row  = lane & 15;
    const int half = lane >> 4;
    const unsigned int* p = (const unsigned int*)(base + row * stride);
#pragma unroll
    for (int v = 0; v < 8; ++v) {
        int k2 = kc + (v >> 2) * 16 + half * 8 + (v & 3) * 2;
        f.u[v] = p[k2 >> 1];
    }
    return f.v;
}

// A fragment straight from an f32 [16][16] LDS tile (K=16 real, padded to 32):
// VGPRs 0..3 cover K=0..15 (real), VGPRs 4..7 cover K=16..31 (zero pad).
__device__ __forceinline__ v16bf load_fragA_f32p16(const float* base, int lane) {
    FragBF f;
    const int row  = lane & 15;
    const int half = lane >> 4;
    const float* p = base + row * 16;
#pragma unroll
    for (int v = 0; v < 4; ++v) {
        int k = half * 8 + v * 2;
        f.u[v] = pk_bf16x2(p[k], p[k + 1]);
    }
#pragma unroll
    for (int v = 4; v < 8; ++v) f.u[v] = 0u;
    return f.v;
}

// B fragment, 32x16 (KxN), W staged transposed WT[n][k]:
// lane n=l&15, half=l>>4; VGPR v: K = (v>>2)*16 + half*8 + (v&3)*2 (+pair).
__device__ __forceinline__ v16bf load_fragB(const __bf16* wtBase, int stride, int kc, int lane) {
    FragBF f;
    const int col  = lane & 15;
    const int half = lane >> 4;
    const unsigned int* p = (const unsigned int*)(wtBase + col * stride);
#pragma unroll
    for (int v = 0; v < 8; ++v) {
        int k2 = kc + (v >> 2) * 16 + half * 8 + (v & 3) * 2;
        f.u[v] = p[k2 >> 1];
    }
    return f.v;
}

// ---------------------------------------------------------------------------
// Generic fused GEMM: out[M,N] = act( (A[M,K] @ W[K,N] + bias) * bnScale + bnBias )
// 128 threads = 4 waves; each wave owns a 16-row tile; WT shared per block.
// Dynamic LDS: (N + 64) * Kpad bf16 elements.
// ---------------------------------------------------------------------------
__global__ void gemm_bias_act(const float* __restrict__ A, const float* __restrict__ W,
                              const float* __restrict__ bias,
                              const float* __restrict__ gamma, const float* __restrict__ beta,
                              float* __restrict__ out, int M, int K, int N, int act) {
    extern __shared__ __align__(16) char smem_raw[];
    __bf16* smem = (__bf16*)smem_raw;
    const int Kpad = (K + 31) & ~31;
    __bf16* WT    = smem;                 // [N][Kpad]
    __bf16* Atile = smem + N * Kpad;      // [4][16][Kpad]

    const int tid  = threadIdx.x;
    const int lane = tid & 31;
    const int wave = tid >> 5;

    for (int idx = tid; idx < N * Kpad; idx += blockDim.x) {
        int n = idx / Kpad, k = idx - n * Kpad;
        WT[idx] = f2bf((k < K) ? W[k * N + n] : 0.0f);
    }
    const int rowbase = (blockIdx.x * 4 + wave) * 16;
    __bf16* Aw = Atile + wave * 16 * Kpad;
    for (int idx = lane; idx < 16 * Kpad; idx += 32) {
        int r = idx / Kpad, k = idx - r * Kpad;
        float v = 0.0f;
        int gr = rowbase + r;
        if (gr < M && k < K) v = A[gr * K + k];
        Aw[idx] = f2bf(v);
    }
    __syncthreads();
    if (rowbase >= M) return;

    const int half = lane >> 4;
    const int col  = lane & 15;
    const float bnk = 0.99999500003749968f;   // 1/sqrt(1 + 1e-5)

    for (int nt = 0; nt < (N >> 4); ++nt) {
        v8f acc = {};
        const __bf16* wtb = WT + (nt * 16) * Kpad;
        for (int kc = 0; kc < Kpad; kc += 32) {
            v16bf a = load_fragA(Aw,  Kpad, kc, lane);
            v16bf b = load_fragB(wtb, Kpad, kc, lane);
            acc = __builtin_amdgcn_wmma_f32_16x16x32_bf16(false, a, false, b,
                                                          (short)0, acc, false, false);
        }
        const int n = nt * 16 + col;
        const float bs  = bias ? bias[n] : 0.0f;
        const float gsc = gamma ? gamma[n] * bnk : 1.0f;
        const float bof = gamma ? beta[n] : 0.0f;
#pragma unroll
        for (int r = 0; r < 8; ++r) {
            int m = rowbase + r + half * 8;
            if (m < M) {
                float v = acc[r] + bs;
                if (gamma) v = v * gsc + bof;
                if (act)   v = fmaxf(v, 0.0f);
                out[m * N + n] = v;
            }
        }
    }
}

// ---------------------------------------------------------------------------
// Fused edge kernel. Per 128-thread block (4 waves x 16 edges):
//  - TDM DMA of the contiguous 64x16 f32 edge_attr slab into LDS (TENSORcnt)
//  - ew^T staged as zero-padded bf16, h[src] rows gathered cooperatively (b128)
//  - msg = relu(h[src] + edge_attr@ew + eb) + eps via WMMA
//  - pass 0: int-bit atomicMax into mmax (msg > 0 so float order == int order)
//  - pass 1: denom += exp(msg-mx); accum += msg*exp(msg-mx)  (L2-resident)
// ---------------------------------------------------------------------------
__global__ void edge_softmax_kernel(const float* __restrict__ edge_attr,
                                    const int*   __restrict__ edge_index,
                                    const float* __restrict__ h,
                                    const float* __restrict__ ew,
                                    const float* __restrict__ eb,
                                    float* __restrict__ mmax,
                                    float* __restrict__ denom,
                                    float* __restrict__ accum,
                                    int E, int pass) {
    __shared__ __align__(16) float  sEA[64 * 16];     // TDM destination (f32)
    __shared__ __align__(16) __bf16 sWT[64 * 32];     // ew^T, K padded 16 -> 32
    __shared__ __align__(16) float  sH[4][16][64];    // gathered h[src] rows
    __shared__ int sSrc[4][16];
    __shared__ int sDst[4][16];

    const int tid  = threadIdx.x;
    const int lane = tid & 31;
    const int wave = tid >> 5;
    const int eblk = blockIdx.x * 64;

    // --- TDM: async DMA of this block's contiguous edge_attr slab into LDS ---
    if (wave == 0) {
        unsigned long long ga =
            (unsigned long long)(uintptr_t)edge_attr + (unsigned long long)eblk * 16ull * 4ull;
        unsigned int ldsa = (unsigned int)(uintptr_t)(&sEA[0]);
        unsigned int td0  = (unsigned int)(E - eblk) * 16u;  // remaining elems; OOB reads -> 0
        u32x4 g0;
        g0.x = 1u;                                            // count=1, user descriptor
        g0.y = ldsa;                                          // lds_addr (bytes)
        g0.z = (unsigned int)(ga & 0xFFFFFFFFull);            // global_addr[31:0]
        g0.w = (unsigned int)((ga >> 32) & 0x01FFFFFFull)     // global_addr[56:32]
             | 0x80000000u;                                   // type=2 ("image")
        i32x8 g1;
        g1[0] = (int)(2u << 16);                              // data_size = 4 bytes
        g1[1] = (int)((td0 & 0xFFFFu) << 16);                 // tensor_dim0[15:0]
        g1[2] = (int)(((td0 >> 16) & 0xFFFFu) | (1u << 16));  // dim0[31:16] | tensor_dim1=1
        g1[3] = (int)(1024u << 16);                           // tile_dim0 = 1024 elems (1-D)
        g1[4] = 0;                                            // tile_dim1/2 = 0 (unused)
        g1[5] = 1024;                                         // tensor_dim0_stride
        g1[6] = 0;
        g1[7] = 0;
        __builtin_amdgcn_tensor_load_to_lds(g0, g1, (i32x4)0, (i32x4)0, (i32x8)0, 0);
    }

    // Stage ew^T as bf16 (zero-pad K 16 -> 32) while the DMA is in flight.
    for (int idx = tid; idx < 64 * 32; idx += blockDim.x) {
        int n = idx >> 5, k = idx & 31;
        sWT[idx] = f2bf((k < 16) ? ew[k * 64 + n] : 0.0f);
    }

    const int ebase = eblk + wave * 16;
    if (lane < 16) {
        int ge = ebase + lane;
        int gc = (ge < E) ? ge : 0;
        sSrc[wave][lane] = edge_index[gc];
        sDst[wave][lane] = edge_index[E + gc];
    }
    // Coalesced gather of the wave's 16 h[src] rows (each 256B contiguous, L2-hot).
    for (int t = lane; t < 16 * 16; t += 32) {
        int r = t >> 4, c4 = t & 15;
        int s = sSrc[wave][r];
        ((float4*)&sH[wave][r][0])[c4] = ((const float4*)(h + (size_t)s * 64))[c4];
    }
    // Speculative prefetch of the next block's slab toward L2.
    if (tid == 0 && eblk + 64 < E)
        __builtin_prefetch(edge_attr + (size_t)(eblk + 64) * 16, 0, 0);

    if (wave == 0) __builtin_amdgcn_s_wait_tensorcnt(0);
    __syncthreads();
    if (ebase >= E) return;

    const int half = lane >> 4;
    const int col  = lane & 15;
    // A fragment is invariant across the 4 output tiles -> build once.
    v16bf a = load_fragA_f32p16(sEA + wave * 16 * 16, lane);

    for (int nt = 0; nt < 4; ++nt) {
        v8f acc = {};
        v16bf b = load_fragB(sWT + (nt * 16) * 32, 32, 0, lane);
        acc = __builtin_amdgcn_wmma_f32_16x16x32_bf16(false, a, false, b,
                                                      (short)0, acc, false, false);
        const int n   = nt * 16 + col;
        const float e = eb[n];
#pragma unroll
        for (int r = 0; r < 8; ++r) {
            int m = r + half * 8;
            int ge = ebase + m;
            if (ge < E) {
                int d = sDst[wave][m];
                float msg = acc[r] + e + sH[wave][m][n];
                msg = fmaxf(msg, 0.0f) + 1e-7f;          // msg > 0 always
                if (pass == 0) {
                    atomicMax((int*)(mmax + d * 64 + n), __float_as_int(msg));
                } else {
                    float mx = mmax[d * 64 + n];
                    float w  = __expf(msg - mx);
                    atomicAdd(denom + d * 64 + n, w);
                    atomicAdd(accum + d * 64 + n, msg * w);
                }
            }
        }
    }
}

// out = accum/denom (0 if no edges) + h    (elementwise, N*64)
__global__ void combine_kernel(const float* __restrict__ accum, const float* __restrict__ denom,
                               const float* __restrict__ h, float* __restrict__ out, int total) {
    int i = blockIdx.x * blockDim.x + threadIdx.x;
    if (i >= total) return;
    float dn = denom[i];
    float a  = (dn > 0.0f) ? (accum[i] / dn) : 0.0f;
    out[i] = a + h[i];
}

// head: out[i] = [h_i (64), gf_i (2)] . head_w + head_b
__global__ void head_kernel(const float* __restrict__ h, const float* __restrict__ gfeat,
                            const int* __restrict__ ngptr,
                            const float* __restrict__ head_w, const float* __restrict__ head_b,
                            float* __restrict__ out, int Nn) {
    int i = blockIdx.x * blockDim.x + threadIdx.x;
    if (i >= Nn) return;
    int G   = ngptr[0];
    int npg = Nn / G;
    int g = i / npg, p = i - g * npg;
    float acc = head_b[0];
    const float* hi = h + i * 64;
#pragma unroll
    for (int f = 0; f < 64; ++f) acc += hi[f] * head_w[f];
    acc += gfeat[g * (2 * npg) + p]       * head_w[64];
    acc += gfeat[g * (2 * npg) + npg + p] * head_w[65];
    out[i] = acc;
}

extern "C" void kernel_launch(void* const* d_in, const int* in_sizes, int n_in,
                              void* d_out, int out_size, void* d_ws, size_t ws_size,
                              hipStream_t stream) {
    (void)n_in; (void)out_size; (void)ws_size;
    const float* x          = (const float*)d_in[0];
    const int*   edge_index = (const int*)  d_in[1];
    const float* edge_attr  = (const float*)d_in[2];
    const int*   ngraphs    = (const int*)  d_in[3];
    const float* gfeat      = (const float*)d_in[4];
    const float* src_w      = (const float*)d_in[5];
    const float* src_b      = (const float*)d_in[6];
    const float* ew[2]  = {(const float*)d_in[7],  (const float*)d_in[15]};
    const float* ebv[2] = {(const float*)d_in[8],  (const float*)d_in[16]};
    const float* w1[2]  = {(const float*)d_in[9],  (const float*)d_in[17]};
    const float* b1[2]  = {(const float*)d_in[10], (const float*)d_in[18]};
    const float* gg[2]  = {(const float*)d_in[11], (const float*)d_in[19]};
    const float* bb[2]  = {(const float*)d_in[12], (const float*)d_in[20]};
    const float* w2[2]  = {(const float*)d_in[13], (const float*)d_in[21]};
    const float* b2[2]  = {(const float*)d_in[14], (const float*)d_in[22]};
    const float* head_w = (const float*)d_in[23];
    const float* head_b = (const float*)d_in[24];
    float* out = (float*)d_out;

    const int NN = in_sizes[0] / 32;   // 100000 nodes
    const int E  = in_sizes[1] / 2;    // 1600000 edges
    const int NH = NN * 64;

    float* ws    = (float*)d_ws;
    float* hA    = ws;                   // [N,64]
    float* hB    = ws + (size_t)NH;      // [N,64]
    float* mmax  = ws + (size_t)2 * NH;
    float* denom = ws + (size_t)3 * NH;
    float* accum = ws + (size_t)4 * NH;
    float* comb  = ws + (size_t)5 * NH;
    float* mid   = ws + (size_t)6 * NH;  // [N,128]

    auto gemmLDS = [](int K, int N) -> size_t {
        int Kp = (K + 31) & ~31;
        return (size_t)(N + 64) * Kp * sizeof(__bf16);
    };

    const int nodeBlocks = (NN + 63) / 64;
    const int edgeBlocks = (E + 63) / 64;

    // Input projection: h = x @ src_w + src_b
    gemm_bias_act<<<nodeBlocks, 128, gemmLDS(32, 64), stream>>>(
        x, src_w, src_b, nullptr, nullptr, hA, NN, 32, 64, 0);

    float* hcur = hA;
    float* hnxt = hB;
    for (int l = 0; l < 2; ++l) {
        // mmax/denom/accum are contiguous -> one memset
        (void)hipMemsetAsync(mmax, 0, (size_t)3 * NH * sizeof(float), stream);
        edge_softmax_kernel<<<edgeBlocks, 128, 0, stream>>>(
            edge_attr, edge_index, hcur, ew[l], ebv[l], mmax, denom, accum, E, 0);
        edge_softmax_kernel<<<edgeBlocks, 128, 0, stream>>>(
            edge_attr, edge_index, hcur, ew[l], ebv[l], mmax, denom, accum, E, 1);
        combine_kernel<<<(NH + 255) / 256, 256, 0, stream>>>(accum, denom, hcur, comb, NH);
        // MLP: Linear(64->128) + BN(eval) + ReLU, then Linear(128->64) + outer ReLU
        gemm_bias_act<<<nodeBlocks, 128, gemmLDS(64, 128), stream>>>(
            comb, w1[l], b1[l], gg[l], bb[l], mid, NN, 64, 128, 1);
        gemm_bias_act<<<nodeBlocks, 128, gemmLDS(128, 64), stream>>>(
            mid, w2[l], b2[l], nullptr, nullptr, hnxt, NN, 128, 64, 1);
        float* t = hcur; hcur = hnxt; hnxt = t;
    }

    head_kernel<<<(NN + 255) / 256, 256, 0, stream>>>(
        hcur, gfeat, ngraphs, head_w, head_b, out, NN);
}
